// ALISTA_AT_40553081209423
// MI455X (gfx1250) — matrix operand order repaired and checked by reading, not compiled, not run on hardware
//
#include <hip/hip_runtime.h>

// ---------------------------------------------------------------------------
// ALISTA on MI455X (gfx1250), bf16 WMMA + async global->LDS staging.
//   r = phi @ x - y^T              (GEMM1)
//   v = x - gamma * (W^T @ r)      (GEMM2, fused epilogue)
//   x = topk-aware soft-threshold  (per-column binary-search select)
//
// GEMM blocking: 128-thread WG = 32 output rows x 256 batch cols.
//   A panel (32 x K, contiguous rows) staged once into LDS via
//   GLOBAL_LOAD_ASYNC_TO_LDS_B128 (padded rows, bank-conflict-free reads).
//   Each wave holds 2 A fragments and register-blocks 2x4 accumulator tiles:
//   every B fragment fetched from L2 feeds two WMMAs (1 b128 load / WMMA).
// ---------------------------------------------------------------------------

typedef unsigned short u16;
typedef __attribute__((ext_vector_type(16))) __bf16 v16bf;
typedef __attribute__((ext_vector_type(8)))  float   v8f;
typedef __attribute__((ext_vector_type(8)))  unsigned short us8;
typedef __attribute__((ext_vector_type(4)))  int     v4i;

typedef v4i __attribute__((address_space(1)))* gv4i_p;   // global int4*
typedef v4i __attribute__((address_space(3)))* lv4i_p;   // LDS int4*

#define MDIM 256
#define NDIM 1024
#define KITER 16
#define BDIM 16384
#define PKEEP 50
#define APAD 8   // u16 pad per LDS row (16B) -> lane l hits bank quad 4l%64

#if __has_builtin(__builtin_amdgcn_global_load_async_to_lds_b128) && \
    __has_builtin(__builtin_amdgcn_s_wait_asynccnt)
#define USE_ASYNC_LDS 1
#else
#define USE_ASYNC_LDS 0
#endif

union FragAB {
    v16bf bf;
    struct { us8 lo, hi; } h;
};

__device__ __forceinline__ u16 f2bf(float f) {
    unsigned u = __float_as_uint(f);
    u += 0x7FFFu + ((u >> 16) & 1u);   // round-to-nearest-even
    return (u16)(u >> 16);
}

// Stage a contiguous (ROWS x K) bf16 row panel into LDS, row stride K+APAD.
template <int K, int ROWS>
__device__ __forceinline__ void stage_a_panel(const u16* __restrict__ A, int row0,
                                              u16* __restrict__ smem)
{
    const int tid = threadIdx.x;
    constexpr int C8    = K / 8;               // b128 chunks per row
    constexpr int ITERS = (ROWS * C8) / 128;
#pragma unroll
    for (int j = 0; j < ITERS; ++j) {
        const int idx = tid + j * 128;
        const int row = idx / C8;
        const int c8  = idx % C8;
        const u16* src = A + (size_t)(row0 + row) * K + c8 * 8;
        u16*       dst = smem + row * (K + APAD) + c8 * 8;
#if USE_ASYNC_LDS
        __builtin_amdgcn_global_load_async_to_lds_b128(
            (gv4i_p)(size_t)src,
            (lv4i_p)(unsigned)(size_t)dst,
            0, 0);
#else
        *(us8*)dst = *(const us8*)src;
#endif
    }
#if USE_ASYNC_LDS
    __builtin_amdgcn_s_wait_asynccnt(0);
#endif
    __syncthreads();
}

// ---------------------------------------------------------------------------
// GEMM1: r_t(B,M) = bf16( phi(M,N) @ x(N,B) - y^T )
// Grid: (M/32) * (B/256) blocks of 128 threads.
// ---------------------------------------------------------------------------
__global__ void __launch_bounds__(128)
alista_gemm1(const u16* __restrict__ phi_bf,   // (M,N) bf16
             const u16* __restrict__ x_bf,     // (B,N) bf16
             const float* __restrict__ y,      // (B,M) f32
             u16* __restrict__ r_t)            // (B,M) bf16
{
    __shared__ __align__(16) u16 sA[32 * (NDIM + APAD)];   // 66 KB

    const int tid   = threadIdx.x;
    const int lane  = tid & 31;
    const int wv    = tid >> 5;
    const int nMT   = MDIM / 32;                  // 8
    const int m0    = (blockIdx.x % nMT) * 32;
    const int bbase = (blockIdx.x / nMT) * 256;

    stage_a_panel<NDIM, 32>(phi_bf, m0, sA);

    const int r15  = lane & 15;
    const int half = lane >> 4;

    const u16* sArow0 = sA + (r15)      * (NDIM + APAD) + (half << 3);
    const u16* sArow1 = sA + (16 + r15) * (NDIM + APAD) + (half << 3);
    const int  b0w    = bbase + wv * 64;
    const u16* bRow0  = x_bf + (size_t)(b0w + r15) * NDIM + (half << 3);

    v8f acc[2][4] = {{{}, {}, {}, {}}, {{}, {}, {}, {}}};
#pragma unroll 1
    for (int k0 = 0; k0 < NDIM; k0 += 32) {
        FragAB a0, a1;
        a0.h.lo = *(const us8*)(sArow0 + k0);
        a0.h.hi = *(const us8*)(sArow0 + k0 + 16);
        a1.h.lo = *(const us8*)(sArow1 + k0);
        a1.h.hi = *(const us8*)(sArow1 + k0 + 16);
        if (k0 + 32 < NDIM)
            __builtin_prefetch(bRow0 + k0 + 32, 0, 1);
#pragma unroll
        for (int t = 0; t < 4; ++t) {
            FragAB b;
            const u16* bp = bRow0 + (size_t)t * 16 * NDIM + k0;
            b.h.lo = *(const us8*)(bp);
            b.h.hi = *(const us8*)(bp + 16);
            acc[0][t] = __builtin_amdgcn_wmma_f32_16x16x32_bf16(
                            false, a0.bf, false, b.bf, (short)0, acc[0][t], false, false);
            acc[1][t] = __builtin_amdgcn_wmma_f32_16x16x32_bf16(
                            false, a1.bf, false, b.bf, (short)0, acc[1][t], false, false);
        }
    }

    // Epilogue: lane -> col b = b0w+t*16+r15, rows m = m0 + u*16 + half*8 + i
#pragma unroll
    for (int u = 0; u < 2; ++u) {
        const int mbase = m0 + u * 16 + (half << 3);
#pragma unroll
        for (int t = 0; t < 4; ++t) {
            const int bb    = b0w + t * 16 + r15;
            const float* yp = y + (size_t)bb * MDIM + mbase;
            us8 o;
#pragma unroll
            for (int i = 0; i < 8; ++i)
                o[i] = f2bf(acc[u][t][i] - yp[i]);
            *(us8*)(r_t + (size_t)bb * MDIM + mbase) = o;
        }
    }
}

// ---------------------------------------------------------------------------
// GEMM2 fused: v_t(B,N) = x_t - gamma[it] * ( W^T(N,M) @ r(M,B) )
// Grid: (N/32) * (B/256) blocks of 128 threads.
// ---------------------------------------------------------------------------
__global__ void __launch_bounds__(128)
alista_gemm2(const u16* __restrict__ wt_bf,    // (N,M) bf16
             const u16* __restrict__ r_t,      // (B,M) bf16
             const float* __restrict__ x_t,    // (B,N) f32
             const float* __restrict__ gamma,  // (K,)
             float* __restrict__ v_t,          // (B,N) f32
             int iter)
{
    __shared__ __align__(16) u16 sA[32 * (MDIM + APAD)];   // 17 KB

    const int tid   = threadIdx.x;
    const int lane  = tid & 31;
    const int wv    = tid >> 5;
    const int nNT   = NDIM / 32;                  // 32
    const int n0    = (blockIdx.x % nNT) * 32;
    const int bbase = (blockIdx.x / nNT) * 256;

    stage_a_panel<MDIM, 32>(wt_bf, n0, sA);

    const int r15  = lane & 15;
    const int half = lane >> 4;

    const u16* sArow0 = sA + (r15)      * (MDIM + APAD) + (half << 3);
    const u16* sArow1 = sA + (16 + r15) * (MDIM + APAD) + (half << 3);
    const int  b0w    = bbase + wv * 64;
    const u16* bRow0  = r_t + (size_t)(b0w + r15) * MDIM + (half << 3);

    v8f acc[2][4] = {{{}, {}, {}, {}}, {{}, {}, {}, {}}};
#pragma unroll 1
    for (int k0 = 0; k0 < MDIM; k0 += 32) {
        FragAB a0, a1;
        a0.h.lo = *(const us8*)(sArow0 + k0);
        a0.h.hi = *(const us8*)(sArow0 + k0 + 16);
        a1.h.lo = *(const us8*)(sArow1 + k0);
        a1.h.hi = *(const us8*)(sArow1 + k0 + 16);
#pragma unroll
        for (int t = 0; t < 4; ++t) {
            FragAB b;
            const u16* bp = bRow0 + (size_t)t * 16 * MDIM + k0;
            b.h.lo = *(const us8*)(bp);
            b.h.hi = *(const us8*)(bp + 16);
            acc[0][t] = __builtin_amdgcn_wmma_f32_16x16x32_bf16(
                            false, a0.bf, false, b.bf, (short)0, acc[0][t], false, false);
            acc[1][t] = __builtin_amdgcn_wmma_f32_16x16x32_bf16(
                            false, a1.bf, false, b.bf, (short)0, acc[1][t], false, false);
        }
    }

    const float gi = gamma[iter];
#pragma unroll
    for (int u = 0; u < 2; ++u) {
        const int nbase = n0 + u * 16 + (half << 3);
#pragma unroll
        for (int t = 0; t < 4; ++t) {
            const int bb    = b0w + t * 16 + r15;
            const float* xp = x_t + (size_t)bb * NDIM + nbase;
            float*       vp = v_t + (size_t)bb * NDIM + nbase;
#pragma unroll
            for (int i = 0; i < 8; ++i)
                vp[i] = xp[i] - gi * acc[u][t][i];
        }
    }
}

// ---------------------------------------------------------------------------
// Per-column top-50 threshold (binary search on float bit pattern) + update:
//   th = theta[it] / (10*|x_old| + 1)
//   x  = (|v| > thr50) ? v : sign(v)*max(|v|-th, 0)
// ---------------------------------------------------------------------------
__global__ void __launch_bounds__(256)
alista_topk_apply(const float* __restrict__ v_t,   // (B,N)
                  float* __restrict__ x_t,         // (B,N) in/out
                  u16* __restrict__ x_bf,          // (B,N) out
                  const float* __restrict__ theta, // (K,)
                  int iter)
{
    const int b   = blockIdx.x;
    const int tid = threadIdx.x;
    const float* vrow = v_t + (size_t)b * NDIM;

    float    v[4];
    unsigned abits[4];
#pragma unroll
    for (int j = 0; j < 4; ++j) {
        v[j]     = vrow[tid + 256 * j];
        abits[j] = __float_as_uint(v[j]) & 0x7FFFFFFFu;
    }

    __shared__ unsigned s_cnt;
    unsigned lo = 0u, hi = 0x7F800000u;
    while (lo < hi) {                       // <= 31 uniform iterations
        unsigned mid = lo + ((hi - lo + 1u) >> 1);
        if (tid == 0) s_cnt = 0u;
        __syncthreads();
        int c = 0;
#pragma unroll
        for (int j = 0; j < 4; ++j) c += (abits[j] >= mid) ? 1 : 0;
        if (c) atomicAdd(&s_cnt, (unsigned)c);
        __syncthreads();
        unsigned total = s_cnt;
        __syncthreads();
        if (total >= PKEEP) lo = mid; else hi = mid - 1u;
    }
    const unsigned thr = lo;   // bit pattern of 50th largest |v|

    const float th_i = theta[iter];
    float* xrow = x_t  + (size_t)b * NDIM;
    u16*   brow = x_bf + (size_t)b * NDIM;
#pragma unroll
    for (int j = 0; j < 4; ++j) {
        const int n = tid + 256 * j;
        const float xo = xrow[n];
        const float th = th_i / (fabsf(xo) * 10.0f + 1.0f);   // theta * g(|x|), EPS=0.1
        const float vv = v[j];
        const float st = copysignf(fmaxf(fabsf(vv) - th, 0.0f), vv);
        const float xn = (abits[j] > thr) ? vv : st;
        xrow[n] = xn;
        brow[n] = f2bf(xn);
    }
}

// ---------------------------------------------------------------------------
__global__ void __launch_bounds__(256)
alista_zero(float* __restrict__ x_t, u16* __restrict__ x_bf,
            float* __restrict__ out_tail)
{
    const int idx = blockIdx.x * 256 + threadIdx.x;
    x_t[idx]  = 0.0f;
    x_bf[idx] = 0;
    if (idx < 2 * KITER) out_tail[idx] = 0.0f;
}

__global__ void __launch_bounds__(256)
alista_convert(const float* __restrict__ phi, const float* __restrict__ W,
               u16* __restrict__ phi_bf, u16* __restrict__ wt_bf)
{
    const int idx = blockIdx.x * 256 + threadIdx.x;   // < M*N
    const int m = idx / NDIM;
    const int n = idx % NDIM;
    phi_bf[idx]         = f2bf(phi[idx]);
    wt_bf[n * MDIM + m] = f2bf(W[idx]);
}

// ---------------------------------------------------------------------------
extern "C" void kernel_launch(void* const* d_in, const int* in_sizes, int n_in,
                              void* d_out, int out_size, void* d_ws, size_t ws_size,
                              hipStream_t stream) {
    (void)in_sizes; (void)n_in; (void)out_size; (void)ws_size;

    const float* y     = (const float*)d_in[0];   // (B,M)
    const float* phi   = (const float*)d_in[1];   // (M,N)
    const float* W     = (const float*)d_in[2];   // (M,N)
    const float* gamma = (const float*)d_in[3];   // (K,)
    const float* theta = (const float*)d_in[4];   // (K,)
    // d_in[5] = info (unused)

    float* x_t      = (float*)d_out;                   // (B,N) == x.T output
    float* out_tail = x_t + (size_t)BDIM * NDIM;       // 2K zeros

    char* ws = (char*)d_ws;
    float* v_t    = (float*)ws;  ws += (size_t)BDIM * NDIM * 4;   // 64 MB
    u16*   x_bf   = (u16*)ws;    ws += (size_t)BDIM * NDIM * 2;   // 32 MB
    u16*   r_t    = (u16*)ws;    ws += (size_t)BDIM * MDIM * 2;   //  8 MB
    u16*   phi_bf = (u16*)ws;    ws += (size_t)MDIM * NDIM * 2;   // .5 MB
    u16*   wt_bf  = (u16*)ws;                                     // .5 MB

    alista_zero<<<(BDIM * NDIM) / 256, 256, 0, stream>>>(x_t, x_bf, out_tail);
    alista_convert<<<(MDIM * NDIM) / 256, 256, 0, stream>>>(phi, W, phi_bf, wt_bf);

    const int g1_blocks = (MDIM / 32) * (BDIM / 256);   // 512
    const int g2_blocks = (NDIM / 32) * (BDIM / 256);   // 2048

    for (int it = 0; it < KITER; ++it) {
        alista_gemm1<<<g1_blocks, 128, 0, stream>>>(phi_bf, x_bf, y, r_t);
        alista_gemm2<<<g2_blocks, 128, 0, stream>>>(wt_bf, r_t, x_t, gamma, v_t, it);
        alista_topk_apply<<<BDIM, 256, 0, stream>>>(v_t, x_t, x_bf, theta, it);
    }
}
